// QUBOGNN_77412490543690
// MI455X (gfx1250) — compile-verified, matrix operand order checked by reading, not run on hardware
//
#include <hip/hip_runtime.h>

// ---------------------------------------------------------------------------
// 2-layer GCN + mean-pool + FC for gfx1250 (MI455X).
// GEMMs use v_wmma_f32_16x16x32_f16 (f16 in, f32 accumulate).
// Aggregation uses global f32 atomics (all data resident in 192MB L2).
// ---------------------------------------------------------------------------

typedef __attribute__((ext_vector_type(16))) _Float16 v16h;
typedef __attribute__((ext_vector_type(8)))  float    v8f;

#define N_NODES   10000
#define N_EDGES   640000
#define F_IN      128
#define H1        64
#define H2        32
#define NCLS      16

// ---------------- precision / prep kernels ----------------

__global__ void k_cast_x(const float* __restrict__ x, _Float16* __restrict__ xh, int n) {
    int i = blockIdx.x * blockDim.x + threadIdx.x;
    if (i < n) xh[i] = (_Float16)x[i];
}

// Transpose W1 [128x64] -> W1T [64x128] f16 ; W2 [64x32] -> W2T [32x64] f16
__global__ void k_prep_w(const float* __restrict__ W1, const float* __restrict__ W2,
                         _Float16* __restrict__ W1T, _Float16* __restrict__ W2T) {
    int i = blockIdx.x * blockDim.x + threadIdx.x;
    if (i < F_IN * H1) { int in = i / H1, o = i % H1; W1T[o * F_IN + in] = (_Float16)W1[i]; }
    if (i < H1 * H2)   { int in = i / H2, o = i % H2; W2T[o * H1  + in] = (_Float16)W2[i]; }
}

__global__ void k_init(float* __restrict__ deg, float* __restrict__ gsum) {
    int i = blockIdx.x * blockDim.x + threadIdx.x;
    if (i < N_NODES) deg[i] = 1.0f;   // self-loop
    if (i < H2)      gsum[i] = 0.0f;
}

__global__ void k_degree(const int* __restrict__ dst, float* __restrict__ deg) {
    int e = blockIdx.x * blockDim.x + threadIdx.x;
    if (e < N_EDGES) atomicAdd(&deg[dst[e]], 1.0f);
}

__global__ void k_dinv(const float* __restrict__ deg, float* __restrict__ dinv) {
    int i = blockIdx.x * blockDim.x + threadIdx.x;
    if (i < N_NODES) dinv[i] = rsqrtf(deg[i]);
}

// ---------------- WMMA GEMM: C[M,N] = A[M,K] * BT[N,K]^T ----------------
// One wave per 16x16 output tile; wave w covers columns [16w,16w+16).
// gridDim.x = M/16 (=625), blockDim.x = 32 * (N/16).
__global__ void k_gemm_wmma(const _Float16* __restrict__ A,
                            const _Float16* __restrict__ BT,
                            float* __restrict__ C, int K, int N) {
    const int lane    = threadIdx.x & 31;
    const int wave    = threadIdx.x >> 5;
    const int rowBase = blockIdx.x * 16;
    const int colBase = wave * 16;

    // 16-bit A-matrix fragment layout (ISA 7.12.2):
    //   lanes 0-15 : row = lane,    halves K = {kb..kb+7, kb+16..kb+23}, kb=0
    //   lanes 16-31: row = lane-16, kb=8
    const int r  = lane & 15;
    const int kb = (lane >> 4) << 3;
    const _Float16* ap = A  + (size_t)(rowBase + r) * K + kb;
    const _Float16* bp = BT + (size_t)(colBase + r) * K + kb;

    v8f acc = {};
    for (int k0 = 0; k0 < K; k0 += 32) {
        union { v16h v; float4 f[2]; } a, b;
        a.f[0] = *(const float4*)(ap + k0);
        a.f[1] = *(const float4*)(ap + k0 + 16);
        b.f[0] = *(const float4*)(bp + k0);
        b.f[1] = *(const float4*)(bp + k0 + 16);
        acc = __builtin_amdgcn_wmma_f32_16x16x32_f16(
            /*neg_a=*/false, a.v, /*neg_b=*/false, b.v,
            /*c_mod=*/(short)0, acc, /*reuse_a=*/false, /*reuse_b=*/false);
    }

    // f32 C/D layout: VGPR rr -> M = rr (lanes 0-15) / rr+8 (lanes 16-31), N = lane%16
    const int n  = lane & 15;
    const int mh = (lane >> 4) << 3;
#pragma unroll
    for (int rr = 0; rr < 8; ++rr)
        C[(size_t)(rowBase + rr + mh) * N + colBase + n] = acc[rr];
}

// ---------------- aggregation ----------------

// agg[n,f] = hlin[n,f] * dinv[n]^2   (self-loop term, also initializes agg)
__global__ void k_self(const float* __restrict__ hlin, const float* __restrict__ dinv,
                       float* __restrict__ agg, int F) {
    int nidx = blockIdx.x, f = threadIdx.x;
    float di = dinv[nidx];
    agg[(size_t)nidx * F + f] = hlin[(size_t)nidx * F + f] * di * di;
}

// agg[dst,f] += hlin[src,f] * dinv[src]*dinv[dst]    (one block per edge)
__global__ void k_scatter(const int* __restrict__ src, const int* __restrict__ dst,
                          const float* __restrict__ dinv, const float* __restrict__ hlin,
                          float* __restrict__ agg, int F) {
    int e = blockIdx.x, f = threadIdx.x;
    int s = src[e], d = dst[e];
    float c = dinv[s] * dinv[d];
    atomicAdd(&agg[(size_t)d * F + f], hlin[(size_t)s * F + f] * c);
}

// h = relu(agg + bias); optionally f32 out and/or f16 out
__global__ void k_postact(const float* __restrict__ agg, const float* __restrict__ bias,
                          float* __restrict__ h, _Float16* __restrict__ hh, int F) {
    int nidx = blockIdx.x, f = threadIdx.x;
    float v = agg[(size_t)nidx * F + f] + bias[f];
    v = v > 0.0f ? v : 0.0f;
    if (h)  h[(size_t)nidx * F + f]  = v;
    if (hh) hh[(size_t)nidx * F + f] = (_Float16)v;
}

// ---------------- mean-pool + FC ----------------

__global__ void k_colsum(const float* __restrict__ h2, float* __restrict__ gsum) {
    int col = threadIdx.x & 31;                      // H2 = 32 columns
    int row = blockIdx.x * 8 + (threadIdx.x >> 5);   // 8 rows per 256-thread block
    float s = 0.0f;
    for (; row < N_NODES; row += gridDim.x * 8)
        s += h2[(size_t)row * H2 + col];
    atomicAdd(&gsum[col], s);
}

__global__ void k_final(const float* __restrict__ gsum, const float* __restrict__ Wfc,
                        const float* __restrict__ bfc, float* __restrict__ out) {
    int c = threadIdx.x;
    if (c < NCLS) {
        float s = bfc[c];
        const float inv_n = 1.0f / (float)N_NODES;
#pragma unroll
        for (int j = 0; j < H2; ++j)
            s += (gsum[j] * inv_n) * Wfc[j * NCLS + c];
        out[c] = s;
    }
}

// ---------------------------------------------------------------------------

extern "C" void kernel_launch(void* const* d_in, const int* in_sizes, int n_in,
                              void* d_out, int out_size, void* d_ws, size_t ws_size,
                              hipStream_t stream) {
    const float* x   = (const float*)d_in[0];   // [10000,128]
    const float* W1  = (const float*)d_in[1];   // [128,64]
    const float* b1  = (const float*)d_in[2];   // [64]
    const float* W2  = (const float*)d_in[3];   // [64,32]
    const float* b2  = (const float*)d_in[4];   // [32]
    const float* Wfc = (const float*)d_in[5];   // [32,16]
    const float* bfc = (const float*)d_in[6];   // [16]
    const int*   ei  = (const int*)d_in[7];     // [2,640000]
    const int* src = ei;
    const int* dst = ei + N_EDGES;
    float* out = (float*)d_out;

    // Workspace carve-out (256B aligned); total ~13 MB.
    char* ws = (char*)d_ws;
    size_t off = 0;
    auto alloc = [&](size_t bytes) -> void* {
        void* p = ws + off;
        off = (off + bytes + 255) & ~(size_t)255;
        return p;
    };
    _Float16* xh    = (_Float16*)alloc((size_t)N_NODES * F_IN * sizeof(_Float16));
    _Float16* W1T   = (_Float16*)alloc((size_t)H1 * F_IN * sizeof(_Float16));
    _Float16* W2T   = (_Float16*)alloc((size_t)H2 * H1 * sizeof(_Float16));
    float*    deg   = (float*)alloc((size_t)N_NODES * sizeof(float));
    float*    dinv  = (float*)alloc((size_t)N_NODES * sizeof(float));
    float*    h1lin = (float*)alloc((size_t)N_NODES * H1 * sizeof(float));
    float*    agg1  = (float*)alloc((size_t)N_NODES * H1 * sizeof(float));
    _Float16* h1h   = (_Float16*)alloc((size_t)N_NODES * H1 * sizeof(_Float16));
    float*    h2lin = (float*)alloc((size_t)N_NODES * H2 * sizeof(float));
    float*    agg2  = (float*)alloc((size_t)N_NODES * H2 * sizeof(float));
    float*    h2    = (float*)alloc((size_t)N_NODES * H2 * sizeof(float));
    float*    gsum  = (float*)alloc((size_t)H2 * sizeof(float));
    (void)ws_size; (void)in_sizes; (void)n_in; (void)out_size;

    // --- prep: f16 casts, transposed weights, degrees ---
    k_cast_x<<<(N_NODES * F_IN + 255) / 256, 256, 0, stream>>>(x, xh, N_NODES * F_IN);
    k_prep_w<<<(F_IN * H1 + 255) / 256, 256, 0, stream>>>(W1, W2, W1T, W2T);
    k_init  <<<(N_NODES + 255) / 256, 256, 0, stream>>>(deg, gsum);
    k_degree<<<(N_EDGES + 255) / 256, 256, 0, stream>>>(dst, deg);
    k_dinv  <<<(N_NODES + 255) / 256, 256, 0, stream>>>(deg, dinv);

    // --- layer 1: h1lin = x @ W1 (WMMA), aggregate, bias+relu -> f16 ---
    k_gemm_wmma<<<N_NODES / 16, 32 * (H1 / 16), 0, stream>>>(xh, W1T, h1lin, F_IN, H1);
    k_self   <<<N_NODES, H1, 0, stream>>>(h1lin, dinv, agg1, H1);
    k_scatter<<<N_EDGES, H1, 0, stream>>>(src, dst, dinv, h1lin, agg1, H1);
    k_postact<<<N_NODES, H1, 0, stream>>>(agg1, b1, (float*)nullptr, h1h, H1);

    // --- layer 2: h2lin = h1 @ W2 (WMMA), aggregate, bias+relu -> f32 ---
    k_gemm_wmma<<<N_NODES / 16, 32 * (H2 / 16), 0, stream>>>(h1h, W2T, h2lin, H1, H2);
    k_self   <<<N_NODES, H2, 0, stream>>>(h2lin, dinv, agg2, H2);
    k_scatter<<<N_EDGES, H2, 0, stream>>>(src, dst, dinv, h2lin, agg2, H2);
    k_postact<<<N_NODES, H2, 0, stream>>>(agg2, b2, h2, (_Float16*)nullptr, H2);

    // --- mean pool + FC ---
    k_colsum<<<100, 256, 0, stream>>>(h2, gsum);
    k_final <<<1, NCLS, 0, stream>>>(gsum, Wfc, bfc, out);
}
